// gru_89283780149391
// MI455X (gfx1250) — compile-verified, matrix operand order checked by reading, not run on hardware
//
#include <hip/hip_runtime.h>
#include <hip/hip_bf16.h>
#include <math.h>

#define T_STEPS 512
#define BATCH   64
#define HID     1024
#define K2      (2*HID)   // 2048, reduction length for both GEMMs

typedef __bf16 bf16;
typedef __attribute__((ext_vector_type(16))) bf16  v16bf;
typedef __attribute__((ext_vector_type(8)))  bf16  v8bf;
typedef __attribute__((ext_vector_type(8)))  float v8f;
typedef __attribute__((ext_vector_type(4)))  float v4f;

#define LDS_STRIDE 40   // 32 k-elems + 8 pad (80B rows: 16B-aligned, bank-conflict-free)
#define NKCHUNKS   (K2/32)

// ---------------------------------------------------------------------------
// Fast activations: v_exp_f32 is exp2 on AMD; v_rcp_f32 for the divide.
// gfx1250 has V_TANH_F32 — use the builtin when the toolchain exposes it.
// ---------------------------------------------------------------------------
__device__ __forceinline__ float fast_sigmoid(float x) {
    return __builtin_amdgcn_rcpf(1.0f + __builtin_amdgcn_exp2f(-1.4426950408889634f * x));
}
__device__ __forceinline__ float fast_tanh(float x) {
#if __has_builtin(__builtin_amdgcn_tanhf)
    return __builtin_amdgcn_tanhf(x);
#else
    // tanh(x) = 2*sigmoid(2x) - 1
    return 2.0f * __builtin_amdgcn_rcpf(1.0f + __builtin_amdgcn_exp2f(-2.8853900817779268f * x)) - 1.0f;
#endif
}

// ---------------------------------------------------------------------------
// Async L2 -> LDS copy (CDNA5 GLOBAL_LOAD_ASYNC_TO_LDS, ASYNCcnt-tracked).
// LDS operand is the low 32 bits of the shared-aperture address
// offset:N applies to BOTH the
// global and LDS addresses per the ISA pseudocode.
// ---------------------------------------------------------------------------
__device__ __forceinline__ void async_chunk_issue(const bf16* __restrict__ gsrc,
                                                  bf16* __restrict__ lbuf) {
    // 128 threads stage a 64n x 32k bf16 chunk (4 KB): 32 B per thread.
    const int n  = threadIdx.x >> 1;
    const int kh = (threadIdx.x & 1) * 16;
    const bf16* g = gsrc + (size_t)n * K2 + kh;
    unsigned l = (unsigned)(uintptr_t)(const void*)&lbuf[n * LDS_STRIDE + kh];
    asm volatile("global_load_async_to_lds_b128 %0, %1, off"
                 :: "v"(l), "v"(g) : "memory");
    asm volatile("global_load_async_to_lds_b128 %0, %1, off offset:16"
                 :: "v"(l), "v"(g) : "memory");
}
__device__ __forceinline__ void wait_async0() {
    asm volatile("s_wait_asynccnt 0x0" ::: "memory");
}

// ---------------------------------------------------------------------------
// One-time (per launch) fp32 -> bf16 weight conversion into workspace.
// ---------------------------------------------------------------------------
__global__ void __launch_bounds__(256)
gru_convert_weights(const float* __restrict__ w1, const float* __restrict__ wh,
                    bf16* __restrict__ w1b, bf16* __restrict__ whb)
{
    const size_t n1 = (size_t)K2 * K2;            // 4M elements
    const size_t tot = n1 + (size_t)HID * K2;     // +2M elements
    for (size_t i = (size_t)blockIdx.x * blockDim.x + threadIdx.x;
         i < tot; i += (size_t)gridDim.x * blockDim.x) {
        if (i < n1) w1b[i] = (bf16)w1[i];
        else        whb[i - n1] = (bf16)wh[i - n1];
    }
}

// ---------------------------------------------------------------------------
// A-operand row pointer: A row b is cat(lo[b,0:H], hi[b,0:H]) in f32.
// ---------------------------------------------------------------------------
__device__ __forceinline__ const float*
arow_ptr(const float* __restrict__ lo, const float* __restrict__ hi,
         int mr, int kb, int kc)
{
    return (kb < HID) ? (lo + (size_t)mr * HID + kb + kc)
                      : (hi + (size_t)mr * HID + (kb - HID) + kc);
}

// ---------------------------------------------------------------------------
// Shared GEMM core: 64(M=batch) x 64(N) tile, K=2048, bf16 WMMA, f32 accum.
// - B (bf16 weights, row-major [N][K]) streams L2 -> LDS via double-buffered
//   GLOBAL_LOAD_ASYNC_TO_LDS; one barrier per K-chunk, prefetch overlaps WMMA.
// - A f32 loads are software-pipelined one chunk ahead (loads issued after
//   the barrier, consumed next iteration -> no loadcnt stall before cvt).
// - All 8 B ds_loads issued before the cvt block so LDS latency hides under
//   the 16 v_cvt_pk_bf16_f32 ops, then 4 WMMAs go back-to-back.
// 4 waves: wave w owns M-rows [16w,16w+16); all waves share the LDS chunk.
// ---------------------------------------------------------------------------
__device__ __forceinline__ void
gemm_accum_64x64(const float* __restrict__ lo, const float* __restrict__ hi,
                 const bf16* __restrict__ wbf, int nbase,
                 bf16* __restrict__ lds0, bf16* __restrict__ lds1, v8f acc[4])
{
    const int lane = threadIdx.x & 31;
    const int wave = threadIdx.x >> 5;
    const int m0   = wave * 16;
    const int mr   = m0 + (lane & 15);        // A-operand row for this lane
    const int kc   = (lane >> 4) * 8;         // A/B k sub-offset per half-wave
    const int nl   = lane & 15;               // B-operand column within tile
    const bf16* wbase = wbf + (size_t)nbase * K2;

    async_chunk_issue(wbase, lds0);           // prefetch weight chunk 0

    // A-operand pipeline prologue: f32 registers for chunk 0
    const float* ar = arow_ptr(lo, hi, mr, 0, kc);
    v4f a0 = *(const v4f*)(ar);
    v4f a1 = *(const v4f*)(ar + 4);
    v4f a2 = *(const v4f*)(ar + 16);
    v4f a3 = *(const v4f*)(ar + 20);

    for (int it = 0; it < NKCHUNKS; ++it) {
        bf16* cur = (it & 1) ? lds1 : lds0;
        bf16* nxt = (it & 1) ? lds0 : lds1;

        wait_async0();        // this wave's in-flight chunk has landed in LDS
        __syncthreads();      // all waves: chunk visible AND prev buffer free

        if (it + 1 < NKCHUNKS)
            async_chunk_issue(wbase + (size_t)(it + 1) * 32, nxt);

        // --- B: issue all four N-tile LDS loads up front ---
        v16bf B[4];
#pragma unroll
        for (int nt = 0; nt < 4; ++nt) {
            const bf16* bp = &cur[(nt * 16 + nl) * LDS_STRIDE + kc];
            v8bf blo = *(const v8bf*)(bp);
            v8bf bhi = *(const v8bf*)(bp + 16);
#pragma unroll
            for (int i = 0; i < 8; ++i) { B[nt][i] = blo[i]; B[nt][8 + i] = bhi[i]; }
        }

        // --- prefetch next chunk's A (f32) while cvt + WMMA run ---
        v4f na0 = a0, na1 = a1, na2 = a2, na3 = a3;
        if (it + 1 < NKCHUNKS) {
            const float* nr = arow_ptr(lo, hi, mr, (it + 1) * 32, kc);
            na0 = *(const v4f*)(nr);
            na1 = *(const v4f*)(nr + 4);
            na2 = *(const v4f*)(nr + 16);
            na3 = *(const v4f*)(nr + 20);
        }

        // --- A operand: 16 f32 -> v16bf (layout per ISA 7.12.2) ---
        v16bf A;
#pragma unroll
        for (int i = 0; i < 4; ++i) {
            A[i]      = (bf16)a0[i];
            A[4 + i]  = (bf16)a1[i];
            A[8 + i]  = (bf16)a2[i];
            A[12 + i] = (bf16)a3[i];
        }

        // --- 4 WMMAs back-to-back ---
#pragma unroll
        for (int nt = 0; nt < 4; ++nt)
            acc[nt] = __builtin_amdgcn_wmma_f32_16x16x32_bf16(
                          false, A, false, B[nt], (short)0, acc[nt], false, false);

        a0 = na0; a1 = na1; a2 = na2; a3 = na3;
    }
}

// ---------------------------------------------------------------------------
// Kernel A: gates = sigmoid(cat(x_t, h) @ W1^T + b1)
//   n <  H : z  -> zbuf[b,n]
//   n >= H : r  -> rhbuf[b,n-H] = r * h[b,n-H]
// grid = 32 blocks (N=2048 / 64), block = 128 threads.
// ---------------------------------------------------------------------------
__global__ void __launch_bounds__(128)
gru_gates_kernel(const float* __restrict__ xt, const float* __restrict__ hprev,
                 const bf16* __restrict__ w1b, const float* __restrict__ b1,
                 float* __restrict__ zbuf, float* __restrict__ rhbuf)
{
    __shared__ __align__(16) bf16 lds[2][64 * LDS_STRIDE];
    const int lane  = threadIdx.x & 31;
    const int wave  = threadIdx.x >> 5;
    const int nbase = blockIdx.x * 64;
    const int m0    = wave * 16;

    v8f acc[4] = {v8f{}, v8f{}, v8f{}, v8f{}};
    gemm_accum_64x64(xt, hprev, w1b, nbase, lds[0], lds[1], acc);

    const bool zside = (nbase < HID);
    const int mhi = (lane >> 4) << 3;
#pragma unroll
    for (int nt = 0; nt < 4; ++nt) {
        const int n = nbase + nt * 16 + (lane & 15);
#pragma unroll
        for (int i = 0; i < 8; ++i) {
            const int m = m0 + i + mhi;                 // batch row
            float s = fast_sigmoid(acc[nt][i] + b1[n]);
            if (zside) {
                zbuf[(size_t)m * HID + n] = s;
            } else {
                const int nn = n - HID;
                rhbuf[(size_t)m * HID + nn] = s * hprev[(size_t)m * HID + nn];
            }
        }
    }
}

// ---------------------------------------------------------------------------
// Kernel B: hg = tanh(cat(r*h, x_t) @ Wh^T + bh); h' = (1-z)h + z*hg
// grid = 16 blocks (N=1024 / 64), block = 128 threads.
// ---------------------------------------------------------------------------
__global__ void __launch_bounds__(128)
gru_update_kernel(const float* __restrict__ xt, const float* __restrict__ hprev,
                  const bf16* __restrict__ whb, const float* __restrict__ bh,
                  const float* __restrict__ zbuf, const float* __restrict__ rhbuf,
                  float* __restrict__ out_t, float* __restrict__ hlast)
{
    __shared__ __align__(16) bf16 lds[2][64 * LDS_STRIDE];
    const int lane  = threadIdx.x & 31;
    const int wave  = threadIdx.x >> 5;
    const int nbase = blockIdx.x * 64;
    const int m0    = wave * 16;

    v8f acc[4] = {v8f{}, v8f{}, v8f{}, v8f{}};
    gemm_accum_64x64(rhbuf, xt, whb, nbase, lds[0], lds[1], acc);  // cat(r*h, x)

    const int mhi = (lane >> 4) << 3;
#pragma unroll
    for (int nt = 0; nt < 4; ++nt) {
        const int n = nbase + nt * 16 + (lane & 15);
#pragma unroll
        for (int i = 0; i < 8; ++i) {
            const int m = m0 + i + mhi;                  // batch row
            const size_t idx = (size_t)m * HID + n;
            float hg = fast_tanh(acc[nt][i] + bh[n]);
            float z  = zbuf[idx];
            float hn = (1.0f - z) * hprev[idx] + z * hg;
            out_t[idx] = hn;
            if (hlast) hlast[idx] = hn;
        }
    }
}

// ---------------------------------------------------------------------------
// Host-side launcher: 1 convert + 2 kernels per timestep, all on `stream`.
// ws layout: [W1 bf16: 8MiB][Wh bf16: 4MiB][zbuf: 256KiB][rhbuf: 256KiB]
// h_{prev} for step t>0 is outputs[t-1] (already in d_out) — no extra state.
// ---------------------------------------------------------------------------
extern "C" void kernel_launch(void* const* d_in, const int* in_sizes, int n_in,
                              void* d_out, int out_size, void* d_ws, size_t ws_size,
                              hipStream_t stream)
{
    const float* input  = (const float*)d_in[0];   // [T,B,H]
    const float* hidden = (const float*)d_in[1];   // [B,H]
    const float* W1     = (const float*)d_in[2];   // [2H,2H]
    const float* b1     = (const float*)d_in[3];   // [2H]
    const float* Wh     = (const float*)d_in[4];   // [H,2H]
    const float* bh     = (const float*)d_in[5];   // [H]
    float* out = (float*)d_out;                    // [T,B,H] ++ [B,H]

    char* ws = (char*)d_ws;
    bf16*  w1b   = (bf16*)ws;
    bf16*  whb   = (bf16*)(ws + (size_t)K2 * K2 * sizeof(bf16));
    float* zbuf  = (float*)(ws + (size_t)K2 * K2 * sizeof(bf16)
                               + (size_t)HID * K2 * sizeof(bf16));
    float* rhbuf = zbuf + (size_t)BATCH * HID;

    gru_convert_weights<<<1536, 256, 0, stream>>>(W1, Wh, w1b, whb);

    for (int t = 0; t < T_STEPS; ++t) {
        const float* xt = input + (size_t)t * BATCH * HID;
        const float* hp = (t == 0) ? hidden : out + (size_t)(t - 1) * BATCH * HID;
        float* ot       = out + (size_t)t * BATCH * HID;
        float* hlast    = (t == T_STEPS - 1) ? out + (size_t)T_STEPS * BATCH * HID
                                             : nullptr;
        gru_gates_kernel<<<dim3(K2 / 64), dim3(128), 0, stream>>>(
            xt, hp, w1b, b1, zbuf, rhbuf);
        gru_update_kernel<<<dim3(HID / 64), dim3(128), 0, stream>>>(
            xt, hp, whb, bh, zbuf, rhbuf, ot, hlast);
    }
}
